// PGExplainer_34342558499150
// MI455X (gfx1250) — matrix operand order repaired and checked by reading, not compiled, not run on hardware
//
#include <hip/hip_runtime.h>
#include <hip/hip_bf16.h>

#define N_NODES 10000
#define N_EDGES 640000
#define D_EMB   128
#define HIDDEN  64

typedef __attribute__((ext_vector_type(16))) __bf16 v16bf;
typedef __attribute__((ext_vector_type(8)))  float  v8f;

// ---------------------------------------------------------------------------
// K0: zero the dense 10000x10000 output (float4 vectorized, grid-stride)
// ---------------------------------------------------------------------------
__global__ void pg_zero_out(float4* __restrict__ out, long long n4) {
    long long i = (long long)blockIdx.x * blockDim.x + threadIdx.x;
    long long stride = (long long)gridDim.x * blockDim.x;
    float4 z = {0.f, 0.f, 0.f, 0.f};
    for (; i < n4; i += stride) out[i] = z;
}

// ---------------------------------------------------------------------------
// K1: P = embed @ W1[0:128,:], Q = embed @ W1[128:256,:]  via bf16 WMMA.
// One wave (32 lanes) computes one 16x16 f32 output tile.
// M tiles = 10000/16 = 625 (exact), N tiles = 64/16 = 4, x2 halves = 5000 tiles.
// ---------------------------------------------------------------------------
__global__ void pg_gemm_pq(const float* __restrict__ embed,
                           const float* __restrict__ W1,
                           float* __restrict__ P,
                           float* __restrict__ Q) {
    const int M_TILES = N_NODES / 16;   // 625
    const int N_TILES = HIDDEN / 16;    // 4
    const int TOTAL   = M_TILES * N_TILES * 2;

    int wave = threadIdx.x >> 5;
    int lane = threadIdx.x & 31;
    int tile = blockIdx.x * (blockDim.x >> 5) + wave;
    if (tile >= TOTAL) return;

    int half = tile / (M_TILES * N_TILES);      // 0 -> P, 1 -> Q
    int rem  = tile % (M_TILES * N_TILES);
    int tm   = rem / N_TILES;
    int tn   = rem % N_TILES;

    int mrow = tm * 16 + (lane & 15);           // A-matrix row for this lane
    int ncol = tn * 16 + (lane & 15);           // B-matrix col for this lane
    int khiA = (lane < 16) ? 0 : 8;             // A: K interleave per half-wave
    int khiB = (lane < 16) ? 0 : 8;             // B: K interleave per half-wave

    const float* arow = embed + (long long)mrow * D_EMB;
    const float* wbase = W1 + (long long)half * 128 * HIDDEN + ncol;

    v8f acc = {};
    for (int kk = 0; kk < D_EMB; kk += 32) {
        v16bf a, b;
#pragma unroll
        for (int v = 0; v < 8; ++v) {
            int kA = kk + ((v >= 4) ? 16 : 0) + khiA + 2 * (v & 3);
            a[2 * v]     = (__bf16)arow[kA];
            a[2 * v + 1] = (__bf16)arow[kA + 1];
            int kB = kk + ((v >= 4) ? 16 : 0) + khiB + 2 * (v & 3);
            b[2 * v]     = (__bf16)wbase[(long long)kB * HIDDEN];
            b[2 * v + 1] = (__bf16)wbase[(long long)(kB + 1) * HIDDEN];
        }
        acc = __builtin_amdgcn_wmma_f32_16x16x32_bf16(
            /*neg_a=*/false, a, /*neg_b=*/false, b,
            /*c_mod=*/(short)0, acc, /*reuse_a=*/false, /*reuse_b=*/false);
    }

    float* out = half ? Q : P;
#pragma unroll
    for (int r = 0; r < 8; ++r) {
        int m = tm * 16 + ((lane < 16) ? r : (8 + r));
        out[(long long)m * HIDDEN + tn * 16 + (lane & 15)] = acc[r];
    }
}

// ---------------------------------------------------------------------------
// K2: per-edge MLP tail + concrete sample. values[e] = 0.5 * gate (pre-scaled
// for the symmetric scatter).
// ---------------------------------------------------------------------------
__device__ __forceinline__ unsigned long long pg_mix64(unsigned long long z) {
    z += 0x9E3779B97F4A7C15ull;
    z = (z ^ (z >> 30)) * 0xBF58476D1CE4E5B9ull;
    z = (z ^ (z >> 27)) * 0x94D049BB133111EBull;
    return z ^ (z >> 31);
}

__global__ void pg_edge_values(const long long* __restrict__ col,
                               const long long* __restrict__ row,
                               const float* __restrict__ P,
                               const float* __restrict__ Q,
                               const float* __restrict__ b1,
                               const float* __restrict__ W2,
                               const float* __restrict__ b2,
                               float* __restrict__ values) {
    int e = blockIdx.x * blockDim.x + threadIdx.x;
    if (e >= N_EDGES) return;
    int c = (int)col[e];
    int r = (int)row[e];

    const float4* p  = (const float4*)(P + (long long)c * HIDDEN);
    const float4* q  = (const float4*)(Q + (long long)r * HIDDEN);
    const float4* bb = (const float4*)b1;
    const float4* w2 = (const float4*)W2;

    float acc = 0.f;
#pragma unroll
    for (int j = 0; j < HIDDEN / 4; ++j) {
        float4 pv = p[j], qv = q[j], bv = bb[j], wv = w2[j];
        acc += fmaxf(pv.x + qv.x + bv.x, 0.f) * wv.x;
        acc += fmaxf(pv.y + qv.y + bv.y, 0.f) * wv.y;
        acc += fmaxf(pv.z + qv.z + bv.z, 0.f) * wv.z;
        acc += fmaxf(pv.w + qv.w + bv.w, 0.f) * wv.w;
    }
    float log_alpha = acc + b2[0];

    // counter-based logistic noise (seed 42); TMP (beta) == 1.0
    unsigned long long h = pg_mix64(((unsigned long long)e << 20) ^ 42ull);
    float u = (float)((h >> 11) + 1ull) * (1.0f / 9007199254740994.0f); // (0,1)
    float noise = logf(u) - log1pf(-u);
    float x = noise + log_alpha;
    float gate = 1.0f / (1.0f + expf(-x));
    values[e] = 0.5f * gate;
}

// ---------------------------------------------------------------------------
// K3..K6: sparse symmetrize + adjacency product, O(E) workspace.
// ---------------------------------------------------------------------------
__global__ void pg_scatter_sym(const long long* __restrict__ col,
                               const long long* __restrict__ row,
                               const float* __restrict__ values,
                               float* __restrict__ out) {
    int e = blockIdx.x * blockDim.x + threadIdx.x;
    if (e >= N_EDGES) return;
    long long c = col[e], r = row[e];
    float v = values[e];
    atomicAdd(out + c * N_NODES + r, v);
    atomicAdd(out + r * N_NODES + c, v);
}

__global__ void pg_gather_sym(const long long* __restrict__ col,
                              const long long* __restrict__ row,
                              const float* __restrict__ out,
                              float* __restrict__ contrib) {
    int e = blockIdx.x * blockDim.x + threadIdx.x;
    if (e >= N_EDGES) return;
    long long c = col[e], r = row[e];
    contrib[e] = out[c * N_NODES + r];
}

__global__ void pg_clear_pos(const long long* __restrict__ col,
                             const long long* __restrict__ row,
                             float* __restrict__ out) {
    int e = blockIdx.x * blockDim.x + threadIdx.x;
    if (e >= N_EDGES) return;
    long long c = col[e], r = row[e];
    out[c * N_NODES + r] = 0.f;   // races write identical zeros: benign
    out[r * N_NODES + c] = 0.f;
}

__global__ void pg_final_scatter(const long long* __restrict__ col,
                                 const long long* __restrict__ row,
                                 const float* __restrict__ contrib,
                                 float* __restrict__ out) {
    int e = blockIdx.x * blockDim.x + threadIdx.x;
    if (e >= N_EDGES) return;
    long long c = col[e], r = row[e];
    // each duplicate edge re-adds sym(c,r) once -> cnt(c,r) * sym(c,r)
    atomicAdd(out + c * N_NODES + r, contrib[e]);
}

// ---------------------------------------------------------------------------
extern "C" void kernel_launch(void* const* d_in, const int* in_sizes, int n_in,
                              void* d_out, int out_size, void* d_ws, size_t ws_size,
                              hipStream_t stream) {
    (void)in_sizes; (void)n_in; (void)out_size; (void)ws_size;

    const float*     embed = (const float*)d_in[0];
    const long long* ei    = (const long long*)d_in[1];
    const float*     W1    = (const float*)d_in[2];
    const float*     b1    = (const float*)d_in[3];
    const float*     W2    = (const float*)d_in[4];
    const float*     b2    = (const float*)d_in[5];
    float*           out   = (float*)d_out;

    const long long* col = ei;            // edge_index[0]
    const long long* row = ei + N_EDGES;  // edge_index[1]

    float* P       = (float*)d_ws;                  // [N_NODES, HIDDEN]
    float* Q       = P + (long long)N_NODES * HIDDEN;
    float* values  = Q + (long long)N_NODES * HIDDEN;
    float* contrib = values + N_EDGES;

    // K0: zero 400MB output (HBM-bound floor ~17us at 23.3 TB/s)
    long long n4 = (long long)N_NODES * N_NODES / 4;
    pg_zero_out<<<4096, 256, 0, stream>>>((float4*)out, n4);

    // K1: WMMA precompute P, Q (5000 tiles, 4 waves per block)
    const int TILES = (N_NODES / 16) * (HIDDEN / 16) * 2;
    pg_gemm_pq<<<(TILES + 3) / 4, 128, 0, stream>>>(embed, W1, P, Q);

    // K2: per-edge gate values
    pg_edge_values<<<(N_EDGES + 255) / 256, 256, 0, stream>>>(
        col, row, P, Q, b1, W2, b2, values);

    // K3..K6: symmetric mask * adjacency, sparse
    const int EB = (N_EDGES + 255) / 256;
    pg_scatter_sym <<<EB, 256, 0, stream>>>(col, row, values, out);
    pg_gather_sym  <<<EB, 256, 0, stream>>>(col, row, out, contrib);
    pg_clear_pos   <<<EB, 256, 0, stream>>>(col, row, out);
    pg_final_scatter<<<EB, 256, 0, stream>>>(col, row, contrib, out);
}
